// HFGATDetailed_84713934946895
// MI455X (gfx1250) — matrix operand order repaired and checked by reading, not compiled, not run on hardware
//
#include <hip/hip_runtime.h>

#define TS 132  // padded LDS tile stride (16x128 fp32 tile)

#define N_ITEMS   100000
#define N_OUTFITS 50000
#define N_USERS   20000

typedef __attribute__((ext_vector_type(16))) __bf16 v16bf;
typedef __attribute__((ext_vector_type(8)))  __bf16 v8bf;
typedef __attribute__((ext_vector_type(8)))  float  v8f;

// ---------------- helpers ----------------

__device__ inline float wave_sum(float v) {
#pragma unroll
    for (int off = 16; off > 0; off >>= 1) v += __shfl_xor(v, off, 32);
    return v;
}

// A-fragment (16x32 bf16): lane holds row m=lane&15, half=lane>>4.
// element e: K = (e&8)*2 + half*8 + (e&7)  -> two contiguous 8-elem runs.
__device__ inline v16bf load_a_frag_bf16(const __bf16* __restrict__ arow, int kb, int half) {
    v8bf lo = *(const v8bf*)(arow + kb + half * 8);
    v8bf hi = *(const v8bf*)(arow + kb + 16 + half * 8);
    v16bf r;
#pragma unroll
    for (int e = 0; e < 8; ++e) { r[e] = lo[e]; r[8 + e] = hi[e]; }
    return r;
}

// B-fragment (32x16 bf16): lane holds col n=lane&15; K = kb + (e&8)*2 + half*8 + (e&7).
// Weights pre-transposed to [N][K] so this is two contiguous 16B reads.
__device__ inline v16bf load_b_frag(const __bf16* __restrict__ brow, int kb, int half) {
    v8bf lo = *(const v8bf*)(brow + kb + half * 8);
    v8bf hi = *(const v8bf*)(brow + kb + 16 + half * 8);
    v16bf r;
#pragma unroll
    for (int e = 0; e < 8; ++e) { r[e] = lo[e]; r[8 + e] = hi[e]; }
    return r;
}

template <int K>
__device__ inline v8f gemm_tile_bf16A(const __bf16* __restrict__ Als,
                                      const __bf16* __restrict__ Bt_tile, int lane) {
    const int half = lane >> 4, mn = lane & 15;
    const __bf16* arow = Als + (size_t)mn * K;
    const __bf16* brow = Bt_tile + (size_t)mn * K;
    v8f acc;
#pragma unroll
    for (int i = 0; i < 8; ++i) acc[i] = 0.f;
#pragma unroll
    for (int kb = 0; kb < K; kb += 32) {
        v16bf af = load_a_frag_bf16(arow, kb, half);
        v16bf bfrag = load_b_frag(brow, kb, half);
        acc = __builtin_amdgcn_wmma_f32_16x16x32_bf16(false, af, false, bfrag,
                                                      (short)0, acc, false, false);
    }
    return acc;
}

// C/D layout: lane holds col n=lane&15; VGPR r -> row r + 8*(lane>>4).
__device__ inline void store_tile(float* __restrict__ tile, int n0, v8f acc,
                                  const float* __restrict__ bias, int lane) {
    const int half = lane >> 4, n = lane & 15;
    const float b = bias[n0 + n];
#pragma unroll
    for (int r = 0; r < 8; ++r) tile[(r + 8 * half) * TS + n0 + n] = acc[r] + b;
}

// Cooperative stage: contiguous 16xK f32 tile (rows consecutive) -> LDS bf16.
template <int K>
__device__ inline void stage_tile_bf16(const float* __restrict__ src, __bf16* __restrict__ dst) {
    constexpr int total2 = 16 * K / 2;  // float2 elements
    const float2* s2 = (const float2*)src;
#pragma unroll 2
    for (int i = threadIdx.x; i < total2; i += 256) {
        float2 f = s2[i];
        dst[2 * i]     = (__bf16)f.x;
        dst[2 * i + 1] = (__bf16)f.y;
    }
}

// ---------------- kernels ----------------

// Wt[n*K + k] = (bf16) W[k*128 + n]   (all weights have N == 128)
__global__ void wtr_kernel(const float* __restrict__ W, __bf16* __restrict__ WT, int K) {
    int idx = blockIdx.x * 256 + threadIdx.x;
    if (idx < K * 128) {
        int n = idx / K, k = idx - n * K;
        WT[idx] = (__bf16)W[k * 128 + n];
    }
}

// Fused: per-modality projection + l2n, concat(384), MLP(LN+ReLU), l2n -> x
__global__ __launch_bounds__(256) void encode_items_kernel(
    const float* __restrict__ img, const float* __restrict__ txt, const float* __restrict__ catx,
    const __bf16* __restrict__ WiT, const float* __restrict__ bi,
    const __bf16* __restrict__ WtT, const float* __restrict__ bt,
    const __bf16* __restrict__ WcT, const float* __restrict__ bc,
    const __bf16* __restrict__ fW1T, const float* __restrict__ fb1,
    const float* __restrict__ fg, const float* __restrict__ fbe,
    const __bf16* __restrict__ fW2T, const float* __restrict__ fb2,
    float* __restrict__ xout) {
    __shared__ float  tile[16 * TS];
    __shared__ __bf16 astage[16 * 768];  // staged A-tile (largest K)
    __shared__ __bf16 xcat[16 * 384];
    __shared__ __bf16 hbuf[16 * 128];
    const int lane = threadIdx.x & 31;
    const int w = threadIdx.x >> 5;
    const int n0 = w * 16;
    const int m0 = blockIdx.x * 16;

    // ---- three projections, each l2-normalized into xcat ----
#pragma unroll
    for (int p = 0; p < 3; ++p) {
        if (p == 0) {
            stage_tile_bf16<512>(img + (size_t)m0 * 512, astage);
            __syncthreads();
            v8f acc = gemm_tile_bf16A<512>(astage, WiT + (size_t)n0 * 512, lane);
            store_tile(tile, n0, acc, bi, lane);
        } else if (p == 1) {
            stage_tile_bf16<768>(txt + (size_t)m0 * 768, astage);
            __syncthreads();
            v8f acc = gemm_tile_bf16A<768>(astage, WtT + (size_t)n0 * 768, lane);
            store_tile(tile, n0, acc, bt, lane);
        } else {
            stage_tile_bf16<64>(catx + (size_t)m0 * 64, astage);
            __syncthreads();
            v8f acc = gemm_tile_bf16A<64>(astage, WcT + (size_t)n0 * 64, lane);
            store_tile(tile, n0, acc, bc, lane);
        }
        __syncthreads();
#pragma unroll
        for (int rr = 0; rr < 2; ++rr) {
            int row = 2 * w + rr;
            float v0 = tile[row * TS + lane],      v1 = tile[row * TS + lane + 32];
            float v2 = tile[row * TS + lane + 64], v3 = tile[row * TS + lane + 96];
            float ss = wave_sum(v0 * v0 + v1 * v1 + v2 * v2 + v3 * v3);
            float sc = 1.f / fmaxf(sqrtf(ss), 1e-12f);
            __bf16* xr = xcat + row * 384 + p * 128;
            xr[lane] = (__bf16)(v0 * sc);      xr[lane + 32] = (__bf16)(v1 * sc);
            xr[lane + 64] = (__bf16)(v2 * sc); xr[lane + 96] = (__bf16)(v3 * sc);
        }
        __syncthreads();
    }

    // ---- fuse GEMM1: xcat(16x384) @ fW1 -> LN -> ReLU -> hbuf ----
    {
        v8f acc = gemm_tile_bf16A<384>(xcat, fW1T + (size_t)n0 * 384, lane);
        store_tile(tile, n0, acc, fb1, lane);
        __syncthreads();
#pragma unroll
        for (int rr = 0; rr < 2; ++rr) {
            int row = 2 * w + rr;
            float v0 = tile[row * TS + lane],      v1 = tile[row * TS + lane + 32];
            float v2 = tile[row * TS + lane + 64], v3 = tile[row * TS + lane + 96];
            float s  = wave_sum(v0 + v1 + v2 + v3);
            float s2 = wave_sum(v0 * v0 + v1 * v1 + v2 * v2 + v3 * v3);
            float mean = s * 0.0078125f;
            float var  = s2 * 0.0078125f - mean * mean;
            float inv  = rsqrtf(var + 1e-5f);
            __bf16* hr = hbuf + row * 128;
            hr[lane]      = (__bf16)fmaxf(fg[lane]      * (v0 - mean) * inv + fbe[lane],      0.f);
            hr[lane + 32] = (__bf16)fmaxf(fg[lane + 32] * (v1 - mean) * inv + fbe[lane + 32], 0.f);
            hr[lane + 64] = (__bf16)fmaxf(fg[lane + 64] * (v2 - mean) * inv + fbe[lane + 64], 0.f);
            hr[lane + 96] = (__bf16)fmaxf(fg[lane + 96] * (v3 - mean) * inv + fbe[lane + 96], 0.f);
        }
        __syncthreads();
    }

    // ---- fuse GEMM2: hbuf @ fW2 -> l2n -> xout ----
    {
        v8f acc = gemm_tile_bf16A<128>(hbuf, fW2T + (size_t)n0 * 128, lane);
        store_tile(tile, n0, acc, fb2, lane);
        __syncthreads();
#pragma unroll
        for (int rr = 0; rr < 2; ++rr) {
            int row = 2 * w + rr;
            float v0 = tile[row * TS + lane],      v1 = tile[row * TS + lane + 32];
            float v2 = tile[row * TS + lane + 64], v3 = tile[row * TS + lane + 96];
            float ss = wave_sum(v0 * v0 + v1 * v1 + v2 * v2 + v3 * v3);
            float sc = 1.f / fmaxf(sqrtf(ss), 1e-12f);
            float* orow = xout + (size_t)(m0 + row) * 128;
            orow[lane] = v0 * sc;      orow[lane + 32] = v1 * sc;
            orow[lane + 64] = v2 * sc; orow[lane + 96] = v3 * sc;
        }
    }
}

// COO SpMM: one wave per edge; lane carries one float4 of the 128-wide row.
__global__ __launch_bounds__(256) void spmm_kernel(
    const int* __restrict__ rows, const int* __restrict__ cols, const float* __restrict__ vals,
    const float* __restrict__ X, float* __restrict__ OUT, int E) {
    int e = blockIdx.x * 8 + (threadIdx.x >> 5);
    if (e >= E) return;
    int lane = threadIdx.x & 31;
    int r = rows[e], c = cols[e];
    float v = vals[e];
    float4 x4 = ((const float4*)(X + (size_t)c * 128))[lane];
    float* dst = OUT + (size_t)r * 128 + lane * 4;
    atomicAdd(dst + 0, v * x4.x);
    atomicAdd(dst + 1, v * x4.y);
    atomicAdd(dst + 2, v * x4.z);
    atomicAdd(dst + 3, v * x4.w);
}

// Fused residual MLP block: out = l2n( res' + (ReLU(LN(agg@W1+b1))@W2 + b2) )
// res' = normalize_res ? l2n(res) : res
__global__ __launch_bounds__(256) void update_kernel(
    const float* __restrict__ AGG, const float* __restrict__ RES, int normalize_res,
    const __bf16* __restrict__ W1T, const float* __restrict__ b1,
    const float* __restrict__ g, const float* __restrict__ be,
    const __bf16* __restrict__ W2T, const float* __restrict__ b2,
    float* __restrict__ OUT) {
    __shared__ float  tile[16 * TS];
    __shared__ __bf16 abuf[16 * 128];
    __shared__ __bf16 hbuf[16 * 128];
    const int lane = threadIdx.x & 31;
    const int w = threadIdx.x >> 5;
    const int n0 = w * 16;
    const int m0 = blockIdx.x * 16;

    stage_tile_bf16<128>(AGG + (size_t)m0 * 128, abuf);
    __syncthreads();

    {   // GEMM1 + LN + ReLU
        v8f acc = gemm_tile_bf16A<128>(abuf, W1T + (size_t)n0 * 128, lane);
        store_tile(tile, n0, acc, b1, lane);
        __syncthreads();
#pragma unroll
        for (int rr = 0; rr < 2; ++rr) {
            int row = 2 * w + rr;
            float v0 = tile[row * TS + lane],      v1 = tile[row * TS + lane + 32];
            float v2 = tile[row * TS + lane + 64], v3 = tile[row * TS + lane + 96];
            float s  = wave_sum(v0 + v1 + v2 + v3);
            float s2 = wave_sum(v0 * v0 + v1 * v1 + v2 * v2 + v3 * v3);
            float mean = s * 0.0078125f;
            float var  = s2 * 0.0078125f - mean * mean;
            float inv  = rsqrtf(var + 1e-5f);
            __bf16* hr = hbuf + row * 128;
            hr[lane]      = (__bf16)fmaxf(g[lane]      * (v0 - mean) * inv + be[lane],      0.f);
            hr[lane + 32] = (__bf16)fmaxf(g[lane + 32] * (v1 - mean) * inv + be[lane + 32], 0.f);
            hr[lane + 64] = (__bf16)fmaxf(g[lane + 64] * (v2 - mean) * inv + be[lane + 64], 0.f);
            hr[lane + 96] = (__bf16)fmaxf(g[lane + 96] * (v3 - mean) * inv + be[lane + 96], 0.f);
        }
        __syncthreads();
    }
    {   // GEMM2 + residual + l2n
        v8f acc = gemm_tile_bf16A<128>(hbuf, W2T + (size_t)n0 * 128, lane);
        store_tile(tile, n0, acc, b2, lane);
        __syncthreads();
#pragma unroll
        for (int rr = 0; rr < 2; ++rr) {
            int row = 2 * w + rr;
            const float* rrow = RES + (size_t)(m0 + row) * 128;
            float r0 = rrow[lane], r1 = rrow[lane + 32], r2 = rrow[lane + 64], r3 = rrow[lane + 96];
            if (normalize_res) {
                float rs = wave_sum(r0 * r0 + r1 * r1 + r2 * r2 + r3 * r3);
                float rc = 1.f / fmaxf(sqrtf(rs), 1e-12f);
                r0 *= rc; r1 *= rc; r2 *= rc; r3 *= rc;
            }
            float t0 = tile[row * TS + lane] + r0,      t1 = tile[row * TS + lane + 32] + r1;
            float t2 = tile[row * TS + lane + 64] + r2, t3 = tile[row * TS + lane + 96] + r3;
            float ss = wave_sum(t0 * t0 + t1 * t1 + t2 * t2 + t3 * t3);
            float sc = 1.f / fmaxf(sqrtf(ss), 1e-12f);
            float* orow = OUT + (size_t)(m0 + row) * 128;
            orow[lane] = t0 * sc;      orow[lane + 32] = t1 * sc;
            orow[lane + 64] = t2 * sc; orow[lane + 96] = t3 * sc;
        }
    }
}

// ---------------- launch ----------------

extern "C" void kernel_launch(void* const* d_in, const int* in_sizes, int n_in,
                              void* d_out, int out_size, void* d_ws, size_t ws_size,
                              hipStream_t stream) {
    (void)n_in; (void)out_size; (void)ws_size;

    const float* image_x = (const float*)d_in[0];
    const float* text_x  = (const float*)d_in[1];
    const float* cat_x   = (const float*)d_in[2];
    const int*   ii_row = (const int*)d_in[3];
    const int*   ii_col = (const int*)d_in[4];
    const float* ii_val = (const float*)d_in[5];
    const int*   oi_row = (const int*)d_in[6];
    const int*   oi_col = (const int*)d_in[7];
    const float* oi_val = (const float*)d_in[8];
    const int*   uo_row = (const int*)d_in[9];
    const int*   uo_col = (const int*)d_in[10];
    const float* uo_val = (const float*)d_in[11];
    const float* Wi = (const float*)d_in[12];  const float* bi = (const float*)d_in[13];
    const float* Wt = (const float*)d_in[14];  const float* bt = (const float*)d_in[15];
    const float* Wc = (const float*)d_in[16];  const float* bc = (const float*)d_in[17];
    const float* fW1 = (const float*)d_in[18]; const float* fb1 = (const float*)d_in[19];
    const float* fg  = (const float*)d_in[20]; const float* fbe = (const float*)d_in[21];
    const float* fW2 = (const float*)d_in[22]; const float* fb2 = (const float*)d_in[23];
    const float* iW1 = (const float*)d_in[24]; const float* ib1 = (const float*)d_in[25];
    const float* ig  = (const float*)d_in[26]; const float* ibe = (const float*)d_in[27];
    const float* iW2 = (const float*)d_in[28]; const float* ib2 = (const float*)d_in[29];
    const float* oW1 = (const float*)d_in[30]; const float* ob1 = (const float*)d_in[31];
    const float* og  = (const float*)d_in[32]; const float* obe = (const float*)d_in[33];
    const float* oW2 = (const float*)d_in[34]; const float* ob2 = (const float*)d_in[35];
    const float* uW1 = (const float*)d_in[36]; const float* ub1 = (const float*)d_in[37];
    const float* ug  = (const float*)d_in[38]; const float* ube = (const float*)d_in[39];
    const float* uW2 = (const float*)d_in[40]; const float* ub2 = (const float*)d_in[41];
    const float* user_base   = (const float*)d_in[42];
    const float* outfit_base = (const float*)d_in[43];

    const int E_ii = in_sizes[3];
    const int E_oi = in_sizes[6];
    const int E_uo = in_sizes[9];

    char* wsb = (char*)d_ws;
    size_t off = 0;
    auto alloc = [&](size_t bytes) -> void* {
        void* p = wsb + off;
        off += (bytes + 255) & ~(size_t)255;
        return p;
    };
    __bf16* WiT  = (__bf16*)alloc((size_t)512 * 128 * 2);
    __bf16* WtT  = (__bf16*)alloc((size_t)768 * 128 * 2);
    __bf16* WcT  = (__bf16*)alloc((size_t)64 * 128 * 2);
    __bf16* fW1T = (__bf16*)alloc((size_t)384 * 128 * 2);
    __bf16* fW2T = (__bf16*)alloc((size_t)128 * 128 * 2);
    __bf16* iW1T = (__bf16*)alloc((size_t)128 * 128 * 2);
    __bf16* iW2T = (__bf16*)alloc((size_t)128 * 128 * 2);
    __bf16* oW1T = (__bf16*)alloc((size_t)128 * 128 * 2);
    __bf16* oW2T = (__bf16*)alloc((size_t)128 * 128 * 2);
    __bf16* uW1T = (__bf16*)alloc((size_t)128 * 128 * 2);
    __bf16* uW2T = (__bf16*)alloc((size_t)128 * 128 * 2);
    float* x     = (float*)alloc((size_t)N_ITEMS   * 128 * 4);
    float* xprop = (float*)alloc((size_t)N_ITEMS   * 128 * 4);
    float* agg   = (float*)alloc((size_t)N_OUTFITS * 128 * 4);
    float* aggu  = (float*)alloc((size_t)N_USERS   * 128 * 4);

    float* user_emb   = (float*)d_out;
    float* outfit_emb = (float*)d_out + (size_t)N_USERS * 128;
    float* item_emb   = (float*)d_out + (size_t)(N_USERS + N_OUTFITS) * 128;

    auto tr = [&](const float* W, __bf16* WT, int K) {
        int total = K * 128;
        wtr_kernel<<<(total + 255) / 256, 256, 0, stream>>>(W, WT, K);
    };
    tr(Wi, WiT, 512); tr(Wt, WtT, 768); tr(Wc, WcT, 64);
    tr(fW1, fW1T, 384); tr(fW2, fW2T, 128);
    tr(iW1, iW1T, 128); tr(iW2, iW2T, 128);
    tr(oW1, oW1T, 128); tr(oW2, oW2T, 128);
    tr(uW1, uW1T, 128); tr(uW2, uW2T, 128);

    hipMemsetAsync(xprop, 0, (size_t)N_ITEMS * 128 * 4, stream);
    hipMemsetAsync(agg,   0, (size_t)N_OUTFITS * 128 * 4, stream);
    hipMemsetAsync(aggu,  0, (size_t)N_USERS * 128 * 4, stream);

    encode_items_kernel<<<N_ITEMS / 16, 256, 0, stream>>>(
        image_x, text_x, cat_x, WiT, bi, WtT, bt, WcT, bc,
        fW1T, fb1, fg, fbe, fW2T, fb2, x);

    spmm_kernel<<<(E_ii + 7) / 8, 256, 0, stream>>>(ii_row, ii_col, ii_val, x, xprop, E_ii);

    update_kernel<<<N_ITEMS / 16, 256, 0, stream>>>(
        xprop, x, 0, iW1T, ib1, ig, ibe, iW2T, ib2, item_emb);

    spmm_kernel<<<(E_oi + 7) / 8, 256, 0, stream>>>(oi_row, oi_col, oi_val, item_emb, agg, E_oi);

    update_kernel<<<N_OUTFITS / 16, 256, 0, stream>>>(
        agg, outfit_base, 1, oW1T, ob1, og, obe, oW2T, ob2, outfit_emb);

    spmm_kernel<<<(E_uo + 7) / 8, 256, 0, stream>>>(uo_row, uo_col, uo_val, outfit_emb, aggu, E_uo);

    update_kernel<<<N_USERS / 16, 256, 0, stream>>>(
        aggu, user_base, 1, uW1T, ub1, ug, ube, uW2T, ub2, user_emb);
}
